// MSARowAttentionWithPairBias_67267777790221
// MI455X (gfx1250) — matrix operand order repaired and checked by reading, not compiled, not run on hardware
//
#include <hip/hip_runtime.h>

typedef __attribute__((ext_vector_type(16))) __bf16 v16bf;
typedef __attribute__((ext_vector_type(8)))  __bf16 v8bf;
typedef __attribute__((ext_vector_type(8)))  float  v8f;

union Frag16 { v16bf v; v8bf h[2]; };
union Pack8  { v8bf v; __bf16 s[8]; };

#define WMMA_BF16(a, b, c) \
  __builtin_amdgcn_wmma_f32_16x16x32_bf16(false, (a), false, (b), (short)0, (c), false, false)

// ---------------- Kernel 1: msa f32 -> bf16 ----------------
__global__ void k_cvt(const float* __restrict__ src, __bf16* __restrict__ dst, int n) {
  int i = blockIdx.x * blockDim.x + threadIdx.x;
  if (i < n) dst[i] = (__bf16)src[i];
}

// ---------------- Kernel 2: weight transpose f32 -> bf16 (256x256) ----------------
__global__ void k_wt(const float* __restrict__ W, __bf16* __restrict__ WT) {
  int o = blockIdx.x;       // output column of W == row of WT
  int i = threadIdx.x;      // input dim
  WT[o * 256 + i] = (__bf16)W[i * 256 + o];
}

// ---------------- Kernel 3: pair bias pb[h,i,j] = pair[i,j,:] @ Wpb[:,h] + bpb[h] ----------------
__global__ void k_pb(const float* __restrict__ pair, const float* __restrict__ Wpb,
                     const float* __restrict__ bpb, float* __restrict__ pb) {
  int ij = blockIdx.x * blockDim.x + threadIdx.x;     // 256*256
  const float* p = pair + (size_t)ij * 128;
  float acc[8];
#pragma unroll
  for (int h = 0; h < 8; ++h) acc[h] = bpb[h];
  for (int c = 0; c < 128; ++c) {
    float pv = p[c];
#pragma unroll
    for (int h = 0; h < 8; ++h) acc[h] += pv * Wpb[c * 8 + h];
  }
#pragma unroll
  for (int h = 0; h < 8; ++h) pb[h * 65536 + ij] = acc[h];
}

// ---------------- Kernel 4: projection GEMM (per n: 256x256 @ 256x256), bf16 WMMA ----------------
// Each wave computes a 16x64 strip (4 accumulators) so the A-fragment is loaded once
// per K-chunk and reused for 4 WMMAs.
// vmode 0: out[(n*256+i)*256 + o] (q/k layout).  vmode 1: vT[((n*8+h)*32+d)*256 + i].
__global__ void __launch_bounds__(256) k_proj(const __bf16* __restrict__ X,
                                              const __bf16* __restrict__ WT,
                                              const float* __restrict__ bias,
                                              __bf16* __restrict__ out,
                                              float scale, int vmode) {
  int wid  = blockIdx.x * 8 + (threadIdx.x >> 5);   // 4096 wave-strips
  int lane = threadIdx.x & 31;
  int n = wid >> 6, rem = wid & 63;
  int it = rem >> 2, og = rem & 3;                  // og: group of 4 output tiles
  int ln = lane & 15, hs = lane >> 4;

  const __bf16* arow = X + (size_t)(n * 256 + it * 16 + ln) * 256;
  const __bf16* brow = WT + (size_t)(og * 64 + ln) * 256;   // +oi*16*256 per tile
  v8f acc[4] = {{}, {}, {}, {}};
#pragma unroll
  for (int kc = 0; kc < 8; ++kc) {
    int ka = kc * 32 + hs * 16;
    Frag16 a;
    a.h[0] = *(const v8bf*)(arow + ka);
    a.h[1] = *(const v8bf*)(arow + ka + 8);
    int kb = kc * 32 + hs * 16;
#pragma unroll
    for (int oi = 0; oi < 4; ++oi) {
      v16bf b = *(const v16bf*)(brow + (size_t)oi * 16 * 256 + kb);
      acc[oi] = WMMA_BF16(a.v, b, acc[oi]);
    }
  }
#pragma unroll
  for (int oi = 0; oi < 4; ++oi) {
    int o = og * 64 + oi * 16 + ln;
    float bv = bias[o];
    if (vmode == 0) {
      __bf16* col = out + (size_t)(n * 256 + it * 16 + hs * 8) * 256 + o;
#pragma unroll
      for (int e = 0; e < 8; ++e) col[e * 256] = (__bf16)((acc[oi][e] + bv) * scale);
    } else {
      int h = o >> 5, d = o & 31;
      Pack8 u;
#pragma unroll
      for (int e = 0; e < 8; ++e) u.s[e] = (__bf16)((acc[oi][e] + bv) * scale);
      *(v8bf*)(out + (size_t)((n * 8 + h) * 32 + d) * 256 + it * 16 + hs * 8) = u.v;
    }
  }
}

// ---------------- Kernel 5: attention per (n, h, i-tile); one wave per block ----------------
__global__ void __launch_bounds__(32) k_attn(const __bf16* __restrict__ q,
                                             const __bf16* __restrict__ kmat,
                                             const __bf16* __restrict__ vT,
                                             const float* __restrict__ pb,
                                             __bf16* __restrict__ ao) {
  __shared__ __bf16 P[16][264];   // padded: conflict-free 16B row reads
  int blk = blockIdx.x;
  int it = blk & 15, nh = blk >> 4;
  int n = nh >> 3, h = nh & 7;
  int lane = threadIdx.x, ln = lane & 15, hs = lane >> 4;

  // Q A-fragment (Hd == 32 == WMMA K: one fragment covers the whole head dim)
  const __bf16* qrow = q + (size_t)(n * 256 + it * 16 + ln) * 256 + h * 32;
  Frag16 qa;
  qa.h[0] = *(const v8bf*)(qrow + hs * 16);
  qa.h[1] = *(const v8bf*)(qrow + hs * 16 + 8);

  // scores strip: 16 x 256 in 16 accumulators (Q fragment reused 16x)
  v8f s[16];
#pragma unroll
  for (int jt = 0; jt < 16; ++jt) {
    const __bf16* krow = kmat + (size_t)(n * 256 + jt * 16 + ln) * 256 + h * 32 + hs * 16;
    v16bf b = *(const v16bf*)krow;
    v8f z = {};
    s[jt] = WMMA_BF16(qa.v, b, z);
  }
  // + pair bias (C layout: element e -> row it*16+hs*8+e, lane&15 -> col jt*16+ln)
  const float* pbase = pb + (size_t)(h * 256 + it * 16 + hs * 8) * 256 + ln;
#pragma unroll
  for (int jt = 0; jt < 16; ++jt)
#pragma unroll
    for (int e = 0; e < 8; ++e) s[jt][e] += pbase[e * 256 + jt * 16];

  // softmax per row: reduce across jt locally, then across the 16-lane half
  float mx[8], inv[8];
#pragma unroll
  for (int e = 0; e < 8; ++e) {
    float m = -3.0e38f;
#pragma unroll
    for (int jt = 0; jt < 16; ++jt) m = fmaxf(m, s[jt][e]);
    for (int d = 1; d < 16; d <<= 1) m = fmaxf(m, __shfl_xor(m, d));
    mx[e] = m;
  }
#pragma unroll
  for (int e = 0; e < 8; ++e) {
    float t = 0.f;
#pragma unroll
    for (int jt = 0; jt < 16; ++jt) { float p = __expf(s[jt][e] - mx[e]); s[jt][e] = p; t += p; }
    for (int d = 1; d < 16; d <<= 1) t += __shfl_xor(t, d);
    inv[e] = 1.f / t;
  }
#pragma unroll
  for (int jt = 0; jt < 16; ++jt)
#pragma unroll
    for (int e = 0; e < 8; ++e)
      P[hs * 8 + e][jt * 16 + ln] = (__bf16)(s[jt][e] * inv[e]);
  __syncthreads();

  // O = P @ V   (V pre-transposed: row d holds 256 contiguous j values)
#pragma unroll
  for (int dt = 0; dt < 2; ++dt) {
    v8f acc = {};
#pragma unroll
    for (int kc = 0; kc < 8; ++kc) {
      Frag16 a;
      const __bf16* prow = &P[ln][kc * 32 + hs * 16];
      a.h[0] = *(const v8bf*)prow;
      a.h[1] = *(const v8bf*)(prow + 8);
      const __bf16* vrow = vT + (size_t)((n * 8 + h) * 32 + dt * 16 + ln) * 256 + kc * 32 + hs * 16;
      v16bf b = *(const v16bf*)vrow;
      acc = WMMA_BF16(a.v, b, acc);
    }
    __bf16* orow = ao + (size_t)(n * 256 + it * 16 + hs * 8) * 256 + h * 32 + dt * 16 + ln;
#pragma unroll
    for (int e = 0; e < 8; ++e) orow[e * 256] = (__bf16)acc[e];
  }
}

// ---------------- Kernel 6: output projection + bias + residual + LayerNorm ----------------
__global__ void __launch_bounds__(256) k_out(const __bf16* __restrict__ A,
                                             const __bf16* __restrict__ WoT,
                                             const float* __restrict__ bo,
                                             const float* __restrict__ msa,
                                             const float* __restrict__ gamma,
                                             const float* __restrict__ beta,
                                             float* __restrict__ out) {
  __shared__ float acc_s[16][264];
  int n = blockIdx.x >> 4, it = blockIdx.x & 15;
  int tid = threadIdx.x;
  int w = tid >> 5, lane = tid & 31;
  int ln = lane & 15, hs = lane >> 4;

  // wave w owns output tiles ot = 2w, 2w+1; A fragment shared between them
  const __bf16* arow = A + (size_t)(n * 256 + it * 16 + ln) * 256;
  const __bf16* brow = WoT + (size_t)(w * 32 + ln) * 256;
  v8f acc[2] = {{}, {}};
#pragma unroll
  for (int kc = 0; kc < 8; ++kc) {
    int ka = kc * 32 + hs * 16;
    Frag16 a;
    a.h[0] = *(const v8bf*)(arow + ka);
    a.h[1] = *(const v8bf*)(arow + ka + 8);
    int kb = kc * 32 + hs * 16;
#pragma unroll
    for (int t = 0; t < 2; ++t) {
      v16bf b = *(const v16bf*)(brow + (size_t)t * 16 * 256 + kb);
      acc[t] = WMMA_BF16(a.v, b, acc[t]);
    }
  }
#pragma unroll
  for (int t = 0; t < 2; ++t) {
    int o = w * 32 + t * 16 + ln;
    float bv = bo[o];
#pragma unroll
    for (int e = 0; e < 8; ++e) acc_s[hs * 8 + e][o] = acc[t][e] + bv;
  }
  __syncthreads();

  // LayerNorm: thread -> row r = tid>>4, contiguous 16-col chunk
  int r = tid >> 4, c0 = (tid & 15) * 16;
  size_t gi = (size_t)(n * 256 + it * 16 + r) * 256;
  float sum = 0.f, sq = 0.f, x[16];
#pragma unroll
  for (int c = 0; c < 16; ++c) {
    float v = msa[gi + c0 + c] + acc_s[r][c0 + c];
    x[c] = v; sum += v; sq += v * v;
  }
  for (int d = 1; d < 16; d <<= 1) { sum += __shfl_xor(sum, d); sq += __shfl_xor(sq, d); }
  float mu = sum * (1.f / 256.f);
  float var = sq * (1.f / 256.f) - mu * mu;
  float rs = rsqrtf(var + 1e-5f);
#pragma unroll
  for (int c = 0; c < 16; ++c)
    out[gi + c0 + c] = (x[c] - mu) * rs * gamma[c0 + c] + beta[c0 + c];
}

extern "C" void kernel_launch(void* const* d_in, const int* in_sizes, int n_in,
                              void* d_out, int out_size, void* d_ws, size_t ws_size,
                              hipStream_t stream) {
  (void)in_sizes; (void)n_in; (void)out_size; (void)ws_size;
  const float* msa  = (const float*)d_in[0];
  const float* pair = (const float*)d_in[1];
  const float* Wq   = (const float*)d_in[2];
  const float* bq   = (const float*)d_in[3];
  const float* Wk   = (const float*)d_in[4];
  const float* bk   = (const float*)d_in[5];
  const float* Wv   = (const float*)d_in[6];
  const float* bv   = (const float*)d_in[7];
  const float* Wpb  = (const float*)d_in[8];
  const float* bpb  = (const float*)d_in[9];
  const float* Wo   = (const float*)d_in[10];
  const float* bo   = (const float*)d_in[11];
  const float* gamma= (const float*)d_in[12];
  const float* beta = (const float*)d_in[13];

  char* ws = (char*)d_ws;
  __bf16* msa_bf = (__bf16*)ws;                                   // 8 MB
  __bf16* WqT    = (__bf16*)(ws + 8388608);                       // 4 x 128 KB
  __bf16* WkT    = WqT + 65536;
  __bf16* WvT    = WkT + 65536;
  __bf16* WoT    = WvT + 65536;
  __bf16* qb     = (__bf16*)(ws + 8388608 + 524288);              // 8 MB
  __bf16* kb     = qb + 4194304;                                  // 8 MB
  __bf16* vT     = kb + 4194304;                                  // 8 MB
  float*  pbw    = (float*)(ws + 8388608 + 524288 + 3 * 8388608); // 2 MB
  __bf16* ao     = (__bf16*)((char*)pbw + 2097152);               // 8 MB

  k_cvt<<<16384, 256, 0, stream>>>(msa, msa_bf, 64 * 256 * 256);
  k_wt<<<256, 256, 0, stream>>>(Wq, WqT);
  k_wt<<<256, 256, 0, stream>>>(Wk, WkT);
  k_wt<<<256, 256, 0, stream>>>(Wv, WvT);
  k_wt<<<256, 256, 0, stream>>>(Wo, WoT);
  k_pb<<<256, 256, 0, stream>>>(pair, Wpb, bpb, pbw);
  k_proj<<<512, 256, 0, stream>>>(msa_bf, WqT, bq, qb, 0.17677669529663687f, 0);
  k_proj<<<512, 256, 0, stream>>>(msa_bf, WkT, bk, kb, 1.0f, 0);
  k_proj<<<512, 256, 0, stream>>>(msa_bf, WvT, bv, vT, 1.0f, 1);
  k_attn<<<8192, 32, 0, stream>>>(qb, kb, vT, pbw, ao);
  k_out<<<1024, 256, 0, stream>>>(ao, WoT, bo, msa, gamma, beta, (float*)d_out);
}
